// ProbabilisticAutoma_3856880631860
// MI455X (gfx1250) — matrix operand order repaired and checked by reading, not compiled, not run on hardware
//
#include <hip/hip_runtime.h>
#include <hip/hip_fp16.h>

typedef __attribute__((ext_vector_type(16))) _Float16 v16h;
typedef __attribute__((ext_vector_type(2)))  _Float16 v2h;
typedef __attribute__((ext_vector_type(8)))  float    v8f;
typedef __attribute__((ext_vector_type(8)))  unsigned v8u;
typedef unsigned int uint;

#define NACT 16
#define NST  64
#define NOUT 2
#define BATCH 4096
#define SEQL  1024

// ---------------------------------------------------------------------------
// State relabeling kappa: physical B-fragment K-slot PK = (f,h,j',p) carries
// logical state  s = 16*(f + 2*(j'>>2)) + 8*h + 2*(j'&3) + p.
// Chosen so that packing two C-fragment VGPRs (rows 16m+8h+{2j,2j+1}) with
// v_cvt_pk_f16_f32 lands directly in B-slot (f=m&1, h, j'=j+4*(m>>1)) of the
// SAME lane-half: the per-step C->B conversion needs no cross-lane traffic.
// ---------------------------------------------------------------------------

// Prep kernel: T^T[a] = transpose(softmax(trans_prob[a]/temp, axis=-1)) in f16,
// emitted in WMMA A-fragment order with K indexed by kappa, so each lane's
// per-step LDS reads are two contiguous ds_load_b128.
//
// A-fragment layout used here, contiguous K (16x32, M x K), fragment = [lane][16 halves]:
//   lanes 0-15  (M=lane):    K {0..15} in slots 0-15
//   lanes 16-31 (M=lane-16): K {16..31} in slots 0-15
__global__ void pa_build_table(const float* __restrict__ trans,
                               const float* __restrict__ temp_p,
                               _Float16* __restrict__ tab) {
  int tid = blockIdx.x * blockDim.x + threadIdx.x;   // 0..1023 == a*64 + s
  if (tid >= NACT * NST) return;
  int a = tid >> 6, s = tid & 63;
  float invt = 1.0f / temp_p[0];
  const float* row = trans + (a * NST + s) * NST;    // logits over t'
  float v[NST];
  float mx = -1e30f;
  #pragma unroll
  for (int t = 0; t < NST; ++t) { v[t] = row[t] * invt; mx = fmaxf(mx, v[t]); }
  float sum = 0.f;
  #pragma unroll
  for (int t = 0; t < NST; ++t) { v[t] = __expf(v[t] - mx); sum += v[t]; }
  float inv = 1.0f / sum;

  // PK = kappa^{-1}(s): which physical K slot carries source-state s.
  int m  = (s >> 4) & 3, hh = (s >> 3) & 1, j = (s >> 1) & 3, p = s & 1;
  int fb = m & 1;
  int jp = j + ((m >> 1) << 2);
  int PK = (fb << 5) | (hh << 4) | (jp << 1) | p;
  // A-fragment placement of K = PK (contiguous: lane-half K>>4, slot K&15):
  int f   = PK >> 5;                           // K-fragment (two 16x32 frags)
  int r   = PK & 31;
  int lhi = (r >> 4) & 1;                      // lane-half carrying this K
  int q   = r & 15;                            // half-slot 0..15 within lane
  #pragma unroll
  for (int t = 0; t < NST; ++t) {              // t' == destination state row M
    int mc = t >> 4;                           // M-chunk
    int l  = (t & 15) | (lhi << 4);            // lane
    int idx = (((a * 4 + mc) * 2 + f) << 9) + l * 16 + q;  // half units
    tab[idx] = (_Float16)(v[t] * inv);
  }
}

// ---------------------------------------------------------------------------
// Main kernel: one wave32 per workgroup, one tile of 16 batch elements.
// State = S^T (64 states x 16 batch columns) held as two WMMA B-fragments in
// VGPRs (lane = batch column). Per step, for each action PRESENT in the tile
// (ballot-skip, uniform branch): C_m = A(a,m,0)xB0 + A(a,m,1)xB1 via
// v_wmma_f32_16x16x32_f16; then a per-lane cndmask selects each column's own
// action; cvt_pkrtz rebuilds B-fragments in-register (no cross-lane moves).
// LDS: 128 KB A-fragment table + 32 KB u16 actions = 160 KB -> two single-wave
// workgroups fit in a 320 KB WGP.
// ---------------------------------------------------------------------------
__global__ __launch_bounds__(32) void pa_scan_kernel(
    const int* __restrict__ act, const float* __restrict__ temp_p,
    const float* __restrict__ fin, const _Float16* __restrict__ tab_g,
    float* __restrict__ out) {
  extern __shared__ char smem[];
  _Float16*       tabs  = (_Float16*)smem;             // 65536 halves = 128 KB
  unsigned short* acts  = (unsigned short*)(smem + 131072);  // 16384 u16 = 32 KB
  const int lane = threadIdx.x;
  const int tile = blockIdx.x;                         // 0..255
  const bool hi  = lane >= 16;

  // Stage transition table into LDS (coalesced b128).
  {
    const uint4* src = (const uint4*)tab_g;            // 8192 uint4
    uint4* dst = (uint4*)tabs;
    for (int i = 0; i < 256; ++i) dst[i * 32 + lane] = src[i * 32 + lane];
  }
  // Stage this tile's actions as LDS[t*16 + m] (u16).
  for (int k = 0; k < 512; ++k) {
    int flat = k * 32 + lane;                          // == m*1024 + t
    int m = flat >> 10, t = flat & 1023;
    acts[t * 16 + m] = (unsigned short)act[(tile * 16 + m) * SEQL + t];
  }
  __syncthreads();

  // s0 one-hot at state 0: kappa^{-1}(0) = PK 0 -> b0[0] low half, lanes 0-15.
  v8u b0 = {0, 0, 0, 0, 0, 0, 0, 0};
  v8u b1 = {0, 0, 0, 0, 0, 0, 0, 0};
  b0[0] = hi ? 0u : 0x3C00u;                           // f16 1.0

  #pragma unroll 1
  for (int t = 0; t < SEQL; ++t) {
    int al = (int)acts[t * 16 + (lane & 15)];          // this column's action
    const v16h B0 = __builtin_bit_cast(v16h, b0);
    const v16h B1 = __builtin_bit_cast(v16h, b1);
    float sel[4][8];
    #pragma unroll
    for (int m = 0; m < 4; ++m)
      #pragma unroll
      for (int r = 0; r < 8; ++r) sel[m][r] = 0.0f;

    #pragma unroll
    for (int a = 0; a < NACT; ++a) {
      // Uniform branch (ballot is scalar): EXEC stays all-ones around WMMA.
      if (__builtin_amdgcn_ballot_w32(al == a)) {
        bool take = (al == a);
        v8f c[4];
        #pragma unroll
        for (int m = 0; m < 4; ++m) {
          const v16h A0 = *reinterpret_cast<const v16h*>(
              tabs + (((a * 4 + m) * 2 + 0) << 9) + lane * 16);
          const v16h A1 = *reinterpret_cast<const v16h*>(
              tabs + (((a * 4 + m) * 2 + 1) << 9) + lane * 16);
          v8f z = {};
          z = __builtin_amdgcn_wmma_f32_16x16x32_f16(
              false, A0, false, B0, (short)0, z, false, false);
          z = __builtin_amdgcn_wmma_f32_16x16x32_f16(
              false, A1, false, B1, (short)0, z, false, false);
          c[m] = z;
        }
        #pragma unroll
        for (int m = 0; m < 4; ++m)
          #pragma unroll
          for (int r = 0; r < 8; ++r) sel[m][r] = take ? c[m][r] : sel[m][r];
      }
    }

    // C-fragments -> B-fragments, pure in-lane thanks to kappa relabeling:
    // pack rows (16m+8h+2j, +1) -> B slot (f=m&1, h, j'=j+4*(m>>1)).
    #pragma unroll
    for (int m = 0; m < 4; ++m) {
      #pragma unroll
      for (int j = 0; j < 4; ++j) {
        uint pk = __builtin_bit_cast(uint, __builtin_amdgcn_cvt_pkrtz(
            sel[m][2 * j], sel[m][2 * j + 1]));
        int jp = j + ((m >> 1) << 2);
        if (m & 1) b1[jp] = pk; else b0[jp] = pk;
      }
    }
  }

  // Epilogue: r[b,:] = s_final[b,:] @ softmax(fin/temp, axis=-1).
  // Lane (half h) slot (f,j') carries states 16*(f+2*(j'>>2)) + 8h + 2*(j'&3)
  // (+1); the other 32 states sit in lane^16 -> one SWAPX16 reduction.
  float invt = 1.0f / temp_p[0];
  float acc0 = 0.f, acc1 = 0.f;
  int h8 = hi ? 8 : 0;
  #pragma unroll
  for (int f = 0; f < 2; ++f) {
    #pragma unroll
    for (int jp = 0; jp < 8; ++jp) {
      uint pk = (f == 0) ? b0[jp] : b1[jp];
      int s = 16 * (f + 2 * (jp >> 2)) + h8 + 2 * (jp & 3);
      union { uint u; _Float16 h[2]; } cv;
      cv.u = pk;
      #pragma unroll
      for (int p = 0; p < 2; ++p) {
        float v = (float)cv.h[p];
        float x0 = fin[(s + p) * NOUT + 0] * invt;
        float x1 = fin[(s + p) * NOUT + 1] * invt;
        float mx = fmaxf(x0, x1);
        float e0 = __expf(x0 - mx), e1 = __expf(x1 - mx);
        float is = 1.0f / (e0 + e1);
        acc0 += v * e0 * is;
        acc1 += v * e1 * is;
      }
    }
  }
  acc0 += __builtin_bit_cast(float,
      __builtin_amdgcn_ds_swizzle(__builtin_bit_cast(int, acc0), 0x401F));
  acc1 += __builtin_bit_cast(float,
      __builtin_amdgcn_ds_swizzle(__builtin_bit_cast(int, acc1), 0x401F));
  if (!hi) {
    int b = tile * 16 + lane;
    out[b * NOUT + 0] = acc0;
    out[b * NOUT + 1] = acc1;
  }
}

extern "C" void kernel_launch(void* const* d_in, const int* in_sizes, int n_in,
                              void* d_out, int out_size, void* d_ws, size_t ws_size,
                              hipStream_t stream) {
  const int*   action_seq = (const int*)d_in[0];
  const float* temp       = (const float*)d_in[1];
  const float* trans      = (const float*)d_in[2];
  const float* fin        = (const float*)d_in[3];
  float* out = (float*)d_out;
  _Float16* tab = (_Float16*)d_ws;               // 128 KB fragment table

  pa_build_table<<<4, 256, 0, stream>>>(trans, temp, tab);
  // 160 KB dynamic LDS per workgroup: two workgroups fit per 320 KB WGP.
  pa_scan_kernel<<<BATCH / 16, 32, 163840, stream>>>(action_seq, temp, fin, tab, out);
}
